// RoadLG_48284022341691
// MI455X (gfx1250) — matrix operand order repaired and checked by reading, not compiled, not run on hardware
//
#include <hip/hip_runtime.h>
#include <hip/hip_bf16.h>
#include <math.h>

typedef __attribute__((ext_vector_type(16))) _Float16     v16h;
typedef __attribute__((ext_vector_type(8)))  float        v8f;
typedef __attribute__((ext_vector_type(4)))  unsigned int v4u;
typedef __attribute__((ext_vector_type(8)))  int          v8i;
typedef __attribute__((ext_vector_type(4)))  int          v4i;

#define NNODES 50000
#define INFEAT 128
#define NEDGES 500000
#define BBATCH 128
#define TSEQ   256
#define DDIM   256
#define NHEAD  8
#define FDIM   32
#define KHOPS  3
#define LSEQ   4
#define DSTATE 8
#define DTRANK 16

#if __has_builtin(__builtin_amdgcn_tensor_load_to_lds)
#define USE_TDM 1
#endif

// ---------------- init: zero denom, per-head tp coefficient, global max ----
__global__ void k0_init(float* denom, float* cvec, float* gmax,
                        const float* __restrict__ W_tp, const float* __restrict__ a_tp) {
  int idx = blockIdx.x * blockDim.x + threadIdx.x;
  if (idx < NNODES * NHEAD) denom[idx] = 0.f;
  if (idx < NHEAD) {
    float s = 0.f;
    for (int f = 0; f < FDIM; ++f) s += W_tp[idx * FDIM + f] * a_tp[idx * FDIM + f];
    cvec[idx] = s;
  }
  if (idx == NHEAD) *gmax = -1e30f;
}

// ---------------- pack weight (KxNc fp32 row-major) into WMMA B fragments --
// layout: Bp[(ntile*(K/32)+ktile)*512 + lane*16 + e]  (halves)
// lane: col = ntile*16 + (lane&15); krow = ktile*32 + (lane>>4)*16 + e
__global__ void kpack_b(const float* __restrict__ W, _Float16* __restrict__ Bp,
                        int K, int Nc) {
  int idx = blockIdx.x * blockDim.x + threadIdx.x;
  if (idx >= K * Nc) return;
  int e     = idx & 15;
  int lane  = (idx >> 4) & 31;
  int chunk = idx >> 9;
  int nkt   = K >> 5;
  int ntile = chunk / nkt, ktile = chunk - ntile * nkt;
  int col   = ntile * 16 + (lane & 15);
  int krow  = ktile * 32 + (lane >> 4) * 16 + e;
  Bp[idx]   = (_Float16)W[(size_t)krow * Nc + col];
}

// ---------------- WMMA GEMM: block = 16 rows x 128 cols, 8 waves -----------
// B panel (8 ntiles x K/32 ktiles x 1KB) staged into LDS via TDM.
// split==1: cols < 256 -> C (ld 256); cols >= 256 -> C2 only for rows t==3
template <int K>
__global__ __launch_bounds__(256) void wmma_gemm2(
    const float* __restrict__ A, const _Float16* __restrict__ Bp,
    const float* __restrict__ bias, float* __restrict__ C,
    float* __restrict__ C2, int M, int Nc, int split)
{
  constexpr int NKT = K >> 5;                    // k-tiles of 32
  __shared__ _Float16 bsh[8 * NKT * 512];        // <= 64 KB

  int nbk  = Nc >> 7;                            // column blocks of 128
  int bm   = blockIdx.x / nbk;
  int bn   = blockIdx.x - bm * nbk;
  int wave = threadIdx.x >> 5;
  int lane = threadIdx.x & 31;

  const _Float16* bsrc = Bp + (size_t)(bn * 8) * NKT * 512;
  const unsigned int panel_bytes = 8u * NKT * 1024u;

#ifdef USE_TDM
  if (threadIdx.x < 32) {                        // wave 0 drives the TDM
    unsigned long long gaddr = (unsigned long long)(const void*)bsrc;
    unsigned int ne8 = panel_bytes >> 3;         // elements of 8 bytes
    v4u g0;
    g0[0] = 1u;                                  // count=1, user mode
    g0[1] = 0u;                                  // lds_addr = 0 (bsh at base)
    g0[2] = (unsigned int)(gaddr & 0xffffffffull);
    g0[3] = (unsigned int)((gaddr >> 32) & 0x1ffffffull) | (2u << 30);  // type=2
    v8i g1;
    g1[0] = (3 << 16);                           // data_size = 3 (8B), no mask
    g1[1] = (int)((ne8 & 0xffffu) << 16);        // tensor_dim0[15:0]
    g1[2] = (int)((ne8 >> 16) & 0xffffu);        // tensor_dim0[31:16]; tensor_dim1=0
    g1[3] = (int)((ne8 & 0xffffu) << 16);        // tile_dim0 = ne8 (<=8192)
    g1[4] = 0;                                   // tile_dim1/2 unused (1-D)
    g1[5] = (int)ne8;                            // tensor_dim0_stride
    g1[6] = 0;
    g1[7] = 0;
    v4i g2 = {0, 0, 0, 0}, g3 = {0, 0, 0, 0};
#if __has_include(<hip/amd_detail/amd_gfx1250_TDM.h>)
    v8i g4 = {0, 0, 0, 0, 0, 0, 0, 0};
    __builtin_amdgcn_tensor_load_to_lds(g0, g1, g2, g3, g4, 0);
#else
    __builtin_amdgcn_tensor_load_to_lds(g0, g1, g2, g3, 0);
#endif
    __builtin_amdgcn_s_wait_tensorcnt(0);
  }
#else
  for (unsigned int i = threadIdx.x; i < (panel_bytes >> 4); i += 256)
    ((uint4*)bsh)[i] = ((const uint4*)bsrc)[i];
#endif
  __syncthreads();

  int sel = lane >> 4;
  int l15 = lane & 15;
  int arow = bm * 16 + l15;
  const float* Arow = A + (size_t)arow * K;
  const _Float16* bw = bsh + (size_t)wave * NKT * 512;

  v8f acc = {0.f, 0.f, 0.f, 0.f, 0.f, 0.f, 0.f, 0.f};
#pragma unroll
  for (int kt = 0; kt < NKT; ++kt) {
    int ka0 = kt * 32 + sel * 16;
    v16h av;
#pragma unroll
    for (int e = 0; e < 8; ++e) av[e]     = (_Float16)Arow[ka0 + e];
#pragma unroll
    for (int e = 0; e < 8; ++e) av[8 + e] = (_Float16)Arow[ka0 + 8 + e];
    v16h bv = *(const v16h*)(bw + (kt * 32 + lane) * 16);   // ds_load_b128 x2
    acc = __builtin_amdgcn_wmma_f32_16x16x32_f16(false, av, false, bv,
                                                 (short)0, acc, false, false);
  }

  int col = bn * 128 + wave * 16 + l15;
#pragma unroll
  for (int i = 0; i < 8; ++i) {
    int row = bm * 16 + sel * 8 + i;
    float v = acc[i];
    if (bias) v += bias[col];
    if (!split) {
      C[(size_t)row * Nc + col] = v;
    } else {
      if (col < DDIM)            C [(size_t)row * DDIM + col] = v;
      else if ((row & 3) == 3)   C2[(size_t)(row >> 2) * DDIM + (col - DDIM)] = v;
    }
  }
}

// ---------------- per-node attention logits --------------------------------
__global__ void k2_node_scores(const float* __restrict__ proj,
                               const float* __restrict__ a_src,
                               const float* __restrict__ a_trg,
                               float* s_src, float* s_trg) {
  int idx = blockIdx.x * blockDim.x + threadIdx.x;
  if (idx >= NNODES * NHEAD) return;
  int n = idx >> 3, h = idx & 7;
  const float* p = proj + (size_t)n * DDIM + h * FDIM;
  float ss = 0.f, st = 0.f;
  for (int f = 0; f < FDIM; ++f) {
    ss += p[f] * a_src[h * FDIM + f];
    st += p[f] * a_trg[h * FDIM + f];
  }
  s_src[idx] = ss;
  s_trg[idx] = st;
}

__device__ inline void atomicMaxF(float* addr, float val) {
  int* ia = (int*)addr;
  int old = *ia;
  while (__int_as_float(old) < val) {
    int prev = atomicCAS(ia, old, __float_as_int(val));
    if (prev == old) break;
    old = prev;
  }
}

// ---------------- edge scores + global max ---------------------------------
__global__ __launch_bounds__(256) void k3_edge_scores(
    const int* __restrict__ eidx, const float* __restrict__ ep,
    const float* __restrict__ s_src, const float* __restrict__ s_trg,
    const float* __restrict__ cvec, float* escore, float* gmax) {
  __shared__ float red[256];
  int e = blockIdx.x * blockDim.x + threadIdx.x;
  float lm = -1e30f;
  if (e < NEDGES) {
    int src = eidx[e], trg = eidx[NEDGES + e];
    float p = ep[e];
    for (int h = 0; h < NHEAD; ++h) {
      float s = s_src[src * NHEAD + h] + s_trg[trg * NHEAD + h] + p * cvec[h];
      s = s > 0.f ? s : 0.2f * s;                       // leaky_relu(0.2)
      escore[(size_t)e * NHEAD + h] = s;
      lm = fmaxf(lm, s);
    }
  }
  red[threadIdx.x] = lm;
  __syncthreads();
  for (int o = 128; o > 0; o >>= 1) {
    if (threadIdx.x < o) red[threadIdx.x] = fmaxf(red[threadIdx.x], red[threadIdx.x + o]);
    __syncthreads();
  }
  if (threadIdx.x == 0) atomicMaxF(gmax, red[0]);
}

// ---------------- exp + denominator ----------------------------------------
__global__ void k4_exp_denom(const int* __restrict__ eidx, float* escore,
                             float* denom, const float* __restrict__ gmax) {
  int e = blockIdx.x * blockDim.x + threadIdx.x;
  if (e >= NEDGES) return;
  int trg = eidx[NEDGES + e];
  float m = *gmax;
  for (int h = 0; h < NHEAD; ++h) {
    float v = expf(escore[(size_t)e * NHEAD + h] - m);
    escore[(size_t)e * NHEAD + h] = v;
    atomicAdd(&denom[trg * NHEAD + h], v);
  }
}

// ---------------- weighted message scatter ---------------------------------
__global__ void k5_messages(const int* __restrict__ eidx, const float* __restrict__ escore,
                            const float* __restrict__ denom, const float* __restrict__ proj,
                            float* gat_acc) {
  int idx = blockIdx.x * blockDim.x + threadIdx.x;   // E*H
  if (idx >= NEDGES * NHEAD) return;
  int e = idx >> 3, h = idx & 7;
  int src = eidx[e], trg = eidx[NEDGES + e];
  float att = escore[idx] / (denom[trg * NHEAD + h] + 1e-16f);
  const float* ps = proj + (size_t)src * DDIM + h * FDIM;
  float* dst = gat_acc + (size_t)trg * DDIM + h * FDIM;
  for (int f = 0; f < FDIM; ++f) atomicAdd(&dst[f], ps[f] * att);
}

// ---------------- bias + ELU; seed seq[:,0,:], zero hop slices -------------
__global__ void k6_finalize(float* gat_acc, const float* __restrict__ gat_bias, float* seq) {
  int idx = blockIdx.x * blockDim.x + threadIdx.x;   // N*256
  if (idx >= NNODES * DDIM) return;
  int n = idx >> 8, c = idx & 255;
  float v = gat_acc[idx] + gat_bias[c];
  v = v > 0.f ? v : expm1f(v);                       // elu
  gat_acc[idx] = v;                                  // becomes nf_gat / nf0
  size_t b = (size_t)n * LSEQ * DDIM + c;
  seq[b] = v;
  seq[b + DDIM] = 0.f;
  seq[b + 2 * DDIM] = 0.f;
  seq[b + 3 * DDIM] = 0.f;
}

// ---------------- sparse hop: seq[:,k,:] += norm * seq[col][k-1] -----------
__global__ void k7_hop(const int* __restrict__ eidx, const float* __restrict__ ep,
                       float* seq, int k) {
  int idx = blockIdx.x * blockDim.x + threadIdx.x;   // E*8 chunks of 32
  if (idx >= NEDGES * 8) return;
  int e = idx >> 3, chunk = idx & 7;
  int row = eidx[e], col = eidx[NEDGES + e];
  float p = ep[e];
  const float* srcp = seq + (size_t)col * LSEQ * DDIM + (k - 1) * DDIM + chunk * 32;
  float* dstp       = seq + (size_t)row * LSEQ * DDIM +  k      * DDIM + chunk * 32;
  for (int f = 0; f < 32; ++f) atomicAdd(&dstp[f], p * srcp[f]);
}

// ---------------- fused mamba tail: conv/silu/x_dbl/dt/scan/gate -----------
__global__ __launch_bounds__(256) void k9_mamba(
    const float* __restrict__ xm, const float* __restrict__ z_last,
    const float* __restrict__ conv_w, const float* __restrict__ conv_b,
    const float* __restrict__ W_x, const float* __restrict__ W_dt,
    const float* __restrict__ b_dt, const float* __restrict__ A_log,
    const float* __restrict__ D_param, float* __restrict__ y_gated)
{
  __shared__ float xc_s[LSEQ][DDIM];
  __shared__ float xd_s[LSEQ][32];
  int n = blockIdx.x;
  int d = threadIdx.x;
  const float* xmn = xm + (size_t)n * LSEQ * DDIM;
  float w0 = conv_w[d*4+0], w1 = conv_w[d*4+1], w2 = conv_w[d*4+2], w3 = conv_w[d*4+3];
  float cb = conv_b[d];
  float xr[LSEQ];
#pragma unroll
  for (int t = 0; t < LSEQ; ++t) xr[t] = xmn[t * DDIM + d];
#pragma unroll
  for (int t = 0; t < LSEQ; ++t) {                  // causal depthwise conv
    float a = cb + w3 * xr[t];
    if (t >= 1) a += w2 * xr[t-1];
    if (t >= 2) a += w1 * xr[t-2];
    if (t >= 3) a += w0 * xr[t-3];
    xc_s[t][d] = a / (1.f + expf(-a));              // silu
  }
  __syncthreads();
  if (d < LSEQ * 32) {                              // x_dbl = xc @ W_x (256x32)
    int t = d >> 5, j = d & 31;
    float s = 0.f;
    for (int q = 0; q < DDIM; ++q) s += xc_s[t][q] * W_x[q * 32 + j];
    xd_s[t][j] = s;
  }
  __syncthreads();
  float Ad[DSTATE], hst[DSTATE];
#pragma unroll
  for (int s = 0; s < DSTATE; ++s) { Ad[s] = -expf(A_log[d * DSTATE + s]); hst[s] = 0.f; }
  float bdt = b_dt[d];
  float ylast = 0.f;
#pragma unroll
  for (int t = 0; t < LSEQ; ++t) {
    float dtv = bdt;
    for (int r = 0; r < DTRANK; ++r) dtv += xd_s[t][r] * W_dt[r * DDIM + d];
    dtv = dtv > 20.f ? dtv : log1pf(expf(dtv));     // softplus
    float xcv = xc_s[t][d];
    float du  = dtv * xcv;
#pragma unroll
    for (int s = 0; s < DSTATE; ++s) {
      float dA = expf(dtv * Ad[s]);
      hst[s] = dA * hst[s] + du * xd_s[t][DTRANK + s];          // B_t
    }
    if (t == LSEQ - 1) {
      float y = 0.f;
#pragma unroll
      for (int s = 0; s < DSTATE; ++s) y += hst[s] * xd_s[t][DTRANK + DSTATE + s];  // C_t
      ylast = y + D_param[d] * xcv;
    }
  }
  float zv = z_last[(size_t)n * DDIM + d];
  y_gated[(size_t)n * DDIM + d] = ylast * (zv / (1.f + expf(-zv)));
}

// ---------------- layernorm(out_last + nf0) --------------------------------
__global__ __launch_bounds__(256) void k11_layernorm(
    const float* __restrict__ out_last, const float* __restrict__ nf0,
    const float* __restrict__ ln_g, const float* __restrict__ ln_b, float* hn)
{
  __shared__ float red[256];
  int n = blockIdx.x, c = threadIdx.x;
  float h = out_last[(size_t)n * DDIM + c] + nf0[(size_t)n * DDIM + c];
  red[c] = h; __syncthreads();
  for (int o = 128; o > 0; o >>= 1) { if (c < o) red[c] += red[c + o]; __syncthreads(); }
  float mu = red[0] * (1.f / DDIM);
  __syncthreads();
  float dv = h - mu;
  red[c] = dv * dv; __syncthreads();
  for (int o = 128; o > 0; o >>= 1) { if (c < o) red[c] += red[c + o]; __syncthreads(); }
  float var = red[0] * (1.f / DDIM);
  hn[(size_t)n * DDIM + c] = dv * rsqrtf(var + 1e-5f) * ln_g[c] + ln_b[c];
}

// ---------------- final gather ---------------------------------------------
__global__ void k13_gather(const int* __restrict__ xids,
                           const float* __restrict__ node_emb, float* out) {
  int i = blockIdx.x;           // 0 .. B*T-1
  int c = threadIdx.x;
  int node = xids[i];
  out[(size_t)i * DDIM + c] = node_emb[(size_t)node * DDIM + c];
}

extern "C" void kernel_launch(void* const* d_in, const int* in_sizes, int n_in,
                              void* d_out, int out_size, void* d_ws, size_t ws_size,
                              hipStream_t stream) {
  (void)in_sizes; (void)n_in; (void)out_size; (void)ws_size;
  const float* nf       = (const float*)d_in[0];
  const int*   eidx     = (const int*)  d_in[1];
  const float* ep       = (const float*)d_in[2];
  const int*   xids     = (const int*)  d_in[3];
  const float* W_proj   = (const float*)d_in[4];
  const float* a_src    = (const float*)d_in[5];
  const float* a_trg    = (const float*)d_in[6];
  const float* W_tp     = (const float*)d_in[7];
  const float* a_tp     = (const float*)d_in[8];
  const float* W_skip   = (const float*)d_in[9];
  const float* gat_bias = (const float*)d_in[10];
  const float* W_in     = (const float*)d_in[11];
  const float* conv_w   = (const float*)d_in[12];
  const float* conv_b   = (const float*)d_in[13];
  const float* W_x      = (const float*)d_in[14];
  const float* W_dt     = (const float*)d_in[15];
  const float* b_dt     = (const float*)d_in[16];
  const float* A_log    = (const float*)d_in[17];
  const float* D_param  = (const float*)d_in[18];
  const float* W_out    = (const float*)d_in[19];
  const float* ln_g     = (const float*)d_in[20];
  const float* ln_b     = (const float*)d_in[21];
  const float* W_o      = (const float*)d_in[22];
  const float* b_o      = (const float*)d_in[23];
  float* outp = (float*)d_out;
  float* ws   = (float*)d_ws;

  size_t o = 0;
  float* proj    = ws + o; o += (size_t)NNODES * DDIM;
  float* gat_acc = ws + o; o += (size_t)NNODES * DDIM;
  float* s_src   = ws + o; o += (size_t)NNODES * NHEAD;
  float* s_trg   = ws + o; o += (size_t)NNODES * NHEAD;
  float* escore  = ws + o; o += (size_t)NEDGES * NHEAD;
  float* denom   = ws + o; o += (size_t)NNODES * NHEAD;
  float* cvec    = ws + o; o += 16;
  float* gmax    = cvec + 8;
  float* seq     = ws + o; o += (size_t)NNODES * LSEQ * DDIM;
  float* xm      = ws + o; o += (size_t)NNODES * LSEQ * DDIM;
  float* z_last  = ws + o; o += (size_t)NNODES * DDIM;
  float* y_gated = ws + o; o += (size_t)NNODES * DDIM;
  // packed f16 weight fragments
  _Float16* bp_proj = (_Float16*)(ws + o); o += (INFEAT * DDIM)     / 2;
  _Float16* bp_skip = (_Float16*)(ws + o); o += (INFEAT * DDIM)     / 2;
  _Float16* bp_in   = (_Float16*)(ws + o); o += (DDIM * 2 * DDIM)   / 2;
  _Float16* bp_out  = (_Float16*)(ws + o); o += (DDIM * DDIM)       / 2;
  _Float16* bp_o    = (_Float16*)(ws + o); o += (DDIM * DDIM)       / 2;

  float* out_last = xm;                              // xm dead after k9
  float* hn       = xm + (size_t)NNODES * DDIM;      // disjoint from out_last
  float* node_emb = proj;                            // proj dead after k5

  dim3 blk(256);

  k0_init<<<(NNODES * NHEAD + 255) / 256, blk, 0, stream>>>(denom, cvec, gmax, W_tp, a_tp);

  // pack all weights into WMMA fragment order (f16)
  kpack_b<<<(INFEAT * DDIM + 255) / 256, blk, 0, stream>>>(W_proj, bp_proj, INFEAT, DDIM);
  kpack_b<<<(INFEAT * DDIM + 255) / 256, blk, 0, stream>>>(W_skip, bp_skip, INFEAT, DDIM);
  kpack_b<<<(DDIM * 2 * DDIM + 255) / 256, blk, 0, stream>>>(W_in, bp_in, DDIM, 2 * DDIM);
  kpack_b<<<(DDIM * DDIM + 255) / 256, blk, 0, stream>>>(W_out, bp_out, DDIM, DDIM);
  kpack_b<<<(DDIM * DDIM + 255) / 256, blk, 0, stream>>>(W_o, bp_o, DDIM, DDIM);

  const int gA  = (NNODES / 16) * (DDIM / 128);            // 6250 blocks
  const int gXZ = (NNODES * LSEQ / 16) * (2 * DDIM / 128); // 50000 blocks

  // GAT projections (WMMA + TDM-staged B panels)
  wmma_gemm2<INFEAT><<<gA, blk, 0, stream>>>(nf, bp_proj, nullptr, proj,    nullptr, NNODES, DDIM, 0);
  wmma_gemm2<INFEAT><<<gA, blk, 0, stream>>>(nf, bp_skip, nullptr, gat_acc, nullptr, NNODES, DDIM, 0);

  k2_node_scores<<<(NNODES * NHEAD + 255) / 256, blk, 0, stream>>>(proj, a_src, a_trg, s_src, s_trg);
  k3_edge_scores<<<(NEDGES + 255) / 256, blk, 0, stream>>>(eidx, ep, s_src, s_trg, cvec, escore, gmax);
  k4_exp_denom  <<<(NEDGES + 255) / 256, blk, 0, stream>>>(eidx, escore, denom, gmax);
  k5_messages   <<<(NEDGES * NHEAD + 255) / 256, blk, 0, stream>>>(eidx, escore, denom, proj, gat_acc);
  k6_finalize   <<<(NNODES * DDIM + 255) / 256, blk, 0, stream>>>(gat_acc, gat_bias, seq);

  for (int k = 1; k <= KHOPS; ++k)
    k7_hop<<<(NEDGES * 8 + 255) / 256, blk, 0, stream>>>(eidx, ep, seq, k);

  // Mamba in-projection (WMMA, split store: xm + z_last)
  wmma_gemm2<DDIM><<<gXZ, blk, 0, stream>>>(seq, bp_in, nullptr, xm, z_last, NNODES * LSEQ, 2 * DDIM, 1);

  k9_mamba<<<NNODES, blk, 0, stream>>>(xm, z_last, conv_w, conv_b, W_x, W_dt, b_dt, A_log, D_param, y_gated);

  // out_last = y_gated @ W_out (WMMA)
  wmma_gemm2<DDIM><<<gA, blk, 0, stream>>>(y_gated, bp_out, nullptr, out_last, nullptr, NNODES, DDIM, 0);

  k11_layernorm<<<NNODES, blk, 0, stream>>>(out_last, gat_acc, ln_g, ln_b, hn);

  // node_emb = hn @ W_o + b_o (WMMA)
  wmma_gemm2<DDIM><<<gA, blk, 0, stream>>>(hn, bp_o, b_o, node_emb, nullptr, NNODES, DDIM, 0);

  k13_gather<<<BBATCH * TSEQ, blk, 0, stream>>>(xids, node_emb, outp);
}
